// KGLayer_59322088292478
// MI455X (gfx1250) — matrix operand order repaired and checked by reading, not compiled, not run on hardware
//
#include <hip/hip_runtime.h>
#include <hip/hip_bf16.h>

typedef __attribute__((ext_vector_type(16))) __bf16 v16bf;
typedef __attribute__((ext_vector_type(8)))  float  v8f;

#define E_EDGES 500000
#define TWO_E   1000000
#define N_ENT   100000
#define N_REL   237
#define IN_DIM  100
#define K_DIM   300
#define KP      320
#define NP      112
#define OUT_DIM 100
#define N_TILES (TWO_E / 16)
#define BN_EPS  1e-5f

// workspace float offsets
#define WS_HSUM  0        // 100
#define WS_HSQ   100      // 100
#define WS_RSQ   200      // 100
#define WS_CSUM  320      // 112
#define WS_CSQ   432      // 112
#define WS_BIASP 1184     // 112
#define WS_SC1   1296     // 112
#define WS_SH1   1408     // 112
#define WS_A2S   1520     // 112
#define WS_A2B   1632     // 1
#define WS_CNT   1640     // 237
#define WS_EBS   1880     // 100000
#define WS_REL   101880   // 23700
#define WS_WBF   125584   // 35840 bf16 = 17920 floats (16-aligned)

__device__ __forceinline__ float eluf(float x) { return x > 0.f ? x : __expf(x) - 1.f; }

__global__ void kzero(float* __restrict__ ws, float* __restrict__ out) {
  size_t nent = (size_t)N_ENT * OUT_DIM;
  size_t total = nent + WS_WBF;
  for (size_t i = blockIdx.x * (size_t)blockDim.x + threadIdx.x; i < total;
       i += (size_t)gridDim.x * blockDim.x) {
    if (i < nent) out[i] = 0.f;
    else ws[i - nent] = 0.f;
  }
}

// per-column sums / sums-of-squares of h, exploiting fwd/inv symmetry:
// cols [0,100)+[100,200): sum over edges of ent[t0]+ent[t1]; cols [200,300): rel, mean 0.
__global__ __launch_bounds__(128) void kstats(const int* __restrict__ trip,
                                              const float* __restrict__ ent,
                                              const float* __restrict__ rel,
                                              float* __restrict__ ws) {
  int f = threadIdx.x;
  bool act = f < IN_DIM;
  float s = 0.f, sq = 0.f, rq = 0.f;
  for (int e = blockIdx.x; e < E_EDGES; e += gridDim.x) {
    int i0 = trip[3 * e], i1 = trip[3 * e + 1], i2 = trip[3 * e + 2];
    if (act) {
      float a = ent[(size_t)i0 * IN_DIM + f];
      float b = ent[(size_t)i1 * IN_DIM + f];
      float r = rel[(size_t)i2 * IN_DIM + f];
      s += a + b; sq += a * a + b * b; rq += r * r;
    }
  }
  if (act) {
    atomicAdd(&ws[WS_HSUM + f], s);
    atomicAdd(&ws[WS_HSQ + f], sq);
    atomicAdd(&ws[WS_RSQ + f], rq);
  }
}

__global__ __launch_bounds__(256) void kcnt(const int* __restrict__ trip, float* __restrict__ ws) {
  __shared__ float bins[N_REL];
  for (int i = threadIdx.x; i < N_REL; i += 256) bins[i] = 0.f;
  __syncthreads();
  for (int e = blockIdx.x * 256 + threadIdx.x; e < E_EDGES; e += gridDim.x * 256)
    atomicAdd(&bins[trip[3 * e + 2]], 1.f);
  __syncthreads();
  for (int i = threadIdx.x; i < N_REL; i += 256) atomicAdd(&ws[WS_CNT + i], bins[i]);
}

// Fold bn0 entirely into the weights/bias:
//   c_pre = Hraw @ (a_W * sc0)^T + (a_b + sh0 @ a_W^T)
// Pack W' = a_W*sc0 as bf16, column-major [NP][KP], zero padded.
__global__ __launch_bounds__(256) void kfold0(const float* __restrict__ aW,
                                              const float* __restrict__ ab,
                                              const float* __restrict__ g0,
                                              const float* __restrict__ b0,
                                              float* __restrict__ ws) {
  __shared__ float ssc[K_DIM], ssh[K_DIM];
  int tid = threadIdx.x;
  const float inv = 1.f / (float)TWO_E;
  for (int f = tid; f < K_DIM; f += 256) {
    float mean, ex2;
    if (f < 200) {
      int ff = (f < 100) ? f : f - 100;
      mean = ws[WS_HSUM + ff] * inv;
      ex2  = ws[WS_HSQ + ff] * inv;
    } else {
      mean = 0.f;
      ex2 = 2.f * ws[WS_RSQ + (f - 200)] * inv;
    }
    float var = ex2 - mean * mean;
    float rs = rsqrtf(var + BN_EPS);
    float sc = g0[f] * rs;
    ssc[f] = sc;
    ssh[f] = b0[f] - mean * sc;
  }
  __syncthreads();
  for (int o = tid; o < NP; o += 256) {
    float b = 0.f;
    if (o < OUT_DIM) {
      b = ab[o];
      for (int f = 0; f < K_DIM; ++f) b += aW[o * K_DIM + f] * ssh[f];
    }
    ws[WS_BIASP + o] = b;
  }
  __bf16* wbf = (__bf16*)(ws + WS_WBF);
  for (int idx = tid; idx < NP * KP; idx += 256) {
    int col = idx / KP, k = idx % KP;
    float v = (col < OUT_DIM && k < K_DIM) ? aW[col * K_DIM + k] * ssc[k] : 0.f;
    wbf[idx] = (__bf16)v;
  }
}

__global__ __launch_bounds__(128) void kfold1(const float* __restrict__ g1,
                                              const float* __restrict__ b1,
                                              const float* __restrict__ a2W,
                                              const float* __restrict__ a2b,
                                              float* __restrict__ ws) {
  __shared__ float ssh1[NP];
  int f = threadIdx.x;
  if (f < NP) {
    float sc = 0.f, sh = 0.f, a2 = 0.f;
    if (f < OUT_DIM) {
      const float inv = 1.f / (float)TWO_E;
      float mu = ws[WS_CSUM + f] * inv;
      float ex2 = ws[WS_CSQ + f] * inv;
      float var = ex2 - mu * mu;
      float rs = rsqrtf(var + BN_EPS);
      sc = g1[f] * rs;
      sh = b1[f] - mu * sc;
      a2 = a2W[f] * sc;
    }
    ws[WS_SC1 + f] = sc; ws[WS_SH1 + f] = sh; ws[WS_A2S + f] = a2;
    ssh1[f] = sh;
  }
  __syncthreads();
  if (f == 0) {
    float acc = a2b[0];
    for (int o = 0; o < OUT_DIM; ++o) acc += a2W[o] * ssh1[o];
    ws[WS_A2B] = acc;
  }
}

// Fused gather + GEMM (bf16 WMMA, f32 acc, bn0 folded into W'). PHASE 0: c stats.
// PHASE 1: bn1 + attention logit + exp + atomic segment scatters.
template <int PHASE>
__global__ __launch_bounds__(256) void kgemm(const int* __restrict__ trip,
                                             const float* __restrict__ ent,
                                             const float* __restrict__ rel,
                                             float* __restrict__ ws,
                                             float* __restrict__ hs) {
  __shared__ __bf16 wsh[NP * KP];
  __shared__ float biassh[NP], sc1sh[NP], sh1sh[NP], a2ssh[NP];

  const int tid = threadIdx.x;
  {
    const unsigned* src = (const unsigned*)(ws + WS_WBF);
    unsigned* dst = (unsigned*)wsh;
    for (int i = tid; i < NP * KP / 2; i += 256) dst[i] = src[i];
    for (int i = tid; i < NP; i += 256) {
      biassh[i] = ws[WS_BIASP + i];
      sc1sh[i] = ws[WS_SC1 + i];
      sh1sh[i] = ws[WS_SH1 + i];
      a2ssh[i] = ws[WS_A2S + i];
    }
  }
  __syncthreads();
  const float a2bt = ws[WS_A2B];
  float* ebs = ws + WS_EBS;
  float* relw = ws + WS_REL;

  const int lane = tid & 31;
  const int wid = tid >> 5;
  const int half = lane >> 4;
  const int colL = lane & 15;
  const int gw = blockIdx.x * 8 + wid;
  const int nw = gridDim.x * 8;

  float cs[7], cq[7];
#pragma unroll
  for (int t = 0; t < 7; ++t) { cs[t] = 0.f; cq[t] = 0.f; }

  for (int tile = gw; tile < N_TILES; tile += nw) {
    // Opaque 32-bit offset (VGPR) defeats LICM of the 70 loop-invariant B-fragment
    // LDS loads (which previously got hoisted + spilled to scratch), while keeping
    // the base in addrspace(3) so they lower to ds_load_b128 (not flat_load).
    unsigned lz = 0;
    asm volatile("" : "+v"(lz));

    // A-matrix row for this lane: row = lane%16 (lanes L and L+16 carry different K chunks)
    int grA = tile * 16 + colL;
    bool fwdA = grA < E_EDGES;
    int eA = fwdA ? grA : grA - E_EDGES;
    int ia = trip[3 * eA], ib = trip[3 * eA + 1];
    const float* p0 = ent + (size_t)(fwdA ? ia : ib) * IN_DIM;
    const float* p1 = ent + (size_t)(fwdA ? ib : ia) * IN_DIM;
    const float* pr = rel + (size_t)trip[3 * eA + 2] * IN_DIM;
    const float sgn = fwdA ? 1.f : -1.f;

    v8f acc[7] = {};

#pragma unroll
    for (int s = 0; s < 10; ++s) {
      const int k0 = s * 32;
      v16bf a;
#pragma unroll
      for (int j = 0; j < 8; ++j) {
        int k = k0 + 16 * half + j;
        float raw = 0.f;
        if (k < K_DIM) raw = (k < 100) ? p0[k] : (k < 200) ? p1[k - 100] : sgn * pr[k - 200];
        a[j] = (__bf16)raw;
      }
#pragma unroll
      for (int j = 0; j < 8; ++j) {
        int k = k0 + 16 * half + 8 + j;
        float raw = 0.f;
        if (k < K_DIM) raw = (k < 100) ? p0[k] : (k < 200) ? p1[k - 100] : sgn * pr[k - 200];
        a[8 + j] = (__bf16)raw;
      }
#pragma unroll
      for (int t = 0; t < 7; ++t) {
        const v16bf b = *reinterpret_cast<const v16bf*>(
            &wsh[lz + (unsigned)((t * 16 + colL) * KP + k0 + 16 * half)]);
        acc[t] = __builtin_amdgcn_wmma_f32_16x16x32_bf16(false, a, false, b, (short)0, acc[t],
                                                         false, false);
      }
    }

    // D layout: VGPR v, lane -> row = v + 8*half, col = t*16 + lane%16
    if (PHASE == 0) {
#pragma unroll
      for (int t = 0; t < 7; ++t) {
        int col = t * 16 + colL;
        if (col < OUT_DIM) {
#pragma unroll
          for (int v = 0; v < 8; ++v) {
            float c = acc[t][v] + biassh[col];
            cs[t] += c; cq[t] += c * c;
          }
        }
      }
    } else {
      float pd[8];
#pragma unroll
      for (int v = 0; v < 8; ++v) pd[v] = 0.f;
#pragma unroll
      for (int t = 0; t < 7; ++t) {
        int col = t * 16 + colL;
        float a2 = a2ssh[col], bi = biassh[col];
#pragma unroll
        for (int v = 0; v < 8; ++v) pd[v] += a2 * (acc[t][v] + bi);
      }
#pragma unroll
      for (int off = 1; off < 16; off <<= 1) {
#pragma unroll
        for (int v = 0; v < 8; ++v) pd[v] += __shfl_xor(pd[v], off, 32);
      }
#pragma unroll
      for (int v = 0; v < 8; ++v) {
        int gr = tile * 16 + half * 8 + v;
        bool fwd = gr < E_EDGES;
        int e = fwd ? gr : gr - E_EDGES;
        int i0 = trip[3 * e], i1 = trip[3 * e + 1], r2 = trip[3 * e + 2];
        int srci = fwd ? i0 : i1;
        float dot = pd[v] + a2bt;
        float bb = dot > 0.f ? -dot : -0.01f * dot;  // -leaky_relu(dot, 0.01)
        float ew = __expf(bb);
        if (colL == 0) atomicAdd(&ebs[srci], ew);  // lanes 0,16 cover the 16 rows
#pragma unroll
        for (int t = 0; t < 7; ++t) {
          int col = t * 16 + colL;
          if (col < OUT_DIM) {
            float cp = acc[t][v] + biassh[col];
            float c = cp * sc1sh[col] + sh1sh[col];
            float tv = ew * c;
            atomicAdd(&hs[(size_t)srci * OUT_DIM + col], tv);
            atomicAdd(&relw[r2 * OUT_DIM + col], fwd ? tv : -tv);
          }
        }
      }
    }
  }

  if (PHASE == 0) {
#pragma unroll
    for (int t = 0; t < 7; ++t) {
      int col = t * 16 + colL;
      if (col < OUT_DIM) {
        atomicAdd(&ws[WS_CSUM + col], cs[t]);
        atomicAdd(&ws[WS_CSQ + col], cq[t]);
      }
    }
  }
}

__global__ void kfinal(const float* __restrict__ ws, float* __restrict__ out) {
  const float* ebs = ws + WS_EBS;
  const float* relw = ws + WS_REL;
  const float* cnt = ws + WS_CNT;
  size_t nent = (size_t)N_ENT * OUT_DIM;
  size_t total = nent + (size_t)N_REL * OUT_DIM;
  for (size_t i = blockIdx.x * (size_t)blockDim.x + threadIdx.x; i < total;
       i += (size_t)gridDim.x * blockDim.x) {
    if (i < nent) {
      float s = ebs[i / OUT_DIM];
      s = (s == 0.f) ? 1e-12f : s;
      out[i] = eluf(out[i] / s);
    } else {
      size_t j = i - nent;
      float cn = cnt[j / OUT_DIM];
      cn = cn < 1.f ? 1.f : cn;
      out[i] = eluf(relw[j] / cn);
    }
  }
}

extern "C" void kernel_launch(void* const* d_in, const int* in_sizes, int n_in,
                              void* d_out, int out_size, void* d_ws, size_t ws_size,
                              hipStream_t stream) {
  const int* trip = (const int*)d_in[0];
  const float* ent = (const float*)d_in[1];
  const float* rel = (const float*)d_in[2];
  const float* aW  = (const float*)d_in[3];
  const float* ab  = (const float*)d_in[4];
  const float* a2W = (const float*)d_in[5];
  const float* a2b = (const float*)d_in[6];
  const float* g0  = (const float*)d_in[7];
  const float* b0  = (const float*)d_in[8];
  const float* g1  = (const float*)d_in[9];
  const float* b1  = (const float*)d_in[10];
  float* out = (float*)d_out;
  float* ws  = (float*)d_ws;

  kzero<<<2048, 256, 0, stream>>>(ws, out);
  kstats<<<512, 128, 0, stream>>>(trip, ent, rel, ws);
  kcnt<<<128, 256, 0, stream>>>(trip, ws);
  kfold0<<<1, 256, 0, stream>>>(aW, ab, g0, b0, ws);
  kgemm<0><<<512, 256, 0, stream>>>(trip, ent, rel, ws, out);
  kfold1<<<1, 128, 0, stream>>>(g1, b1, a2W, a2b, ws);
  kgemm<1><<<512, 256, 0, stream>>>(trip, ent, rel, ws, out);
  kfinal<<<4096, 256, 0, stream>>>(ws, out);
}
